// MultiheadAttention_29111288332961
// MI455X (gfx1250) — compile-verified, matrix operand order checked by reading, not run on hardware
//
#include <hip/hip_runtime.h>
#include <hip/hip_bf16.h>

// Problem constants (from reference setup_inputs)
#define BB    4
#define LQ    1024
#define LC    2047
#define LKK   2048     // Lc + 1
#define FD    2048
#define LG2FD 11       // log2(FD)
#define NH    16
#define HD    128
#define HHALF 64

typedef __bf16 bf16x16 __attribute__((ext_vector_type(16)));
typedef __bf16 bf16x4  __attribute__((ext_vector_type(4)));
typedef float  f32x8   __attribute__((ext_vector_type(8)));

// CDNA5 16-bit A/B fragment K-index mapping (CDNA5 ISA §7.12.2):
// lanes 0-15 hold K in {0..7, 16..23}, lanes 16-31 hold K in {8..15, 24..31}
__device__ __forceinline__ int kmap(int i, int hi) {
  return ((i < 8) ? i : i + 8) + hi * 8;
}

__device__ __forceinline__ f32x8 wmma_bf16(bf16x16 a, bf16x16 b, f32x8 c) {
  // (neg_a, A, neg_b, B, c_mod, C, reuse_a, reuse_b)
  return __builtin_amdgcn_wmma_f32_16x16x32_bf16(false, a, false, b, (short)0, c,
                                                 false, false);
}

// ---------------------------------------------------------------------------
// 1) key_new / val_new = kv[:, -1, :] @ Wk / Wv   (tiny: 4 x 2048 rows)
// ---------------------------------------------------------------------------
__global__ __launch_bounds__(256) void kv_proj_kernel(
    const float* __restrict__ kv, const float* __restrict__ Wk,
    const float* __restrict__ Wv, float* __restrict__ kn, float* __restrict__ vn) {
  unsigned idx = blockIdx.x * 256u + threadIdx.x;
  if (idx >= BB * FD) return;
  unsigned b = idx >> LG2FD, f = idx & (FD - 1);
  const float* x = kv + (b << LG2FD);
  float sk = 0.f, sv = 0.f;
  for (unsigned k = 0; k < FD; ++k) {
    float xv = x[k];
    sk += xv * Wk[(k << LG2FD) + f];
    sv += xv * Wv[(k << LG2FD) + f];
  }
  kn[idx] = sk;
  vn[idx] = sv;
}

// ---------------------------------------------------------------------------
// 2) Single pass over the 134MB cache: concat cache+new token into
//    kv_cache_out (f32, d_out region), RoPE'd K (bf16, ws) and a
//    deinterleaved bf16 V plane (ws).  One thread per rotation pair.
// ---------------------------------------------------------------------------
__global__ __launch_bounds__(256) void concat_rope_k_kernel(
    const float* __restrict__ kv_cache, const float* __restrict__ kn,
    const float* __restrict__ vn, float* __restrict__ kv_out,
    __bf16* __restrict__ Krope, __bf16* __restrict__ Vsep) {
  unsigned idx = blockIdx.x * 256u + threadIdx.x;
  if (idx >= (unsigned)BB * LKK * (FD / 2)) return;
  unsigned p = idx & (FD / 2 - 1);
  unsigned bt = idx >> (LG2FD - 1);
  unsigned t = bt & (LKK - 1);
  unsigned b = bt >> 11;          // log2(LKK) = 11
  unsigned h = p >> 6, j = p & 63;
  unsigned f0 = h * HD + j, f1 = f0 + HHALF;

  float k0, k1, v0, v1;
  if (t < LC) {
    const float* src = kv_cache + 2u * ((b * LC + t) << LG2FD);
    k0 = src[f0 * 2 + 0]; v0 = src[f0 * 2 + 1];
    k1 = src[f1 * 2 + 0]; v1 = src[f1 * 2 + 1];
  } else {
    k0 = kn[(b << LG2FD) + f0]; v0 = vn[(b << LG2FD) + f0];
    k1 = kn[(b << LG2FD) + f1]; v1 = vn[(b << LG2FD) + f1];
  }
  float* dst = kv_out + 2u * (bt << LG2FD);
  dst[f0 * 2 + 0] = k0; dst[f0 * 2 + 1] = v0;
  dst[f1 * 2 + 0] = k1; dst[f1 * 2 + 1] = v1;

  // RoPE, base 1000: theta_j = 1000^(-2j/d), angle = t * theta_j
  float theta = __powf(1000.0f, -((float)(2 * j) / (float)HD));
  float s, c;
  __sincosf((float)t * theta, &s, &c);
  unsigned kb = bt << LG2FD;
  Krope[kb + f0] = (__bf16)(k0 * c - k1 * s);
  Krope[kb + f1] = (__bf16)(k1 * c + k0 * s);
  Vsep[kb + f0] = (__bf16)v0;
  Vsep[kb + f1] = (__bf16)v1;
}

// ---------------------------------------------------------------------------
// 3) RoPE on projected Q (positions 0..Lq-1): f32 in -> bf16 out
// ---------------------------------------------------------------------------
__global__ __launch_bounds__(256) void rope_q_kernel(
    const float* __restrict__ Qraw, __bf16* __restrict__ Qrope) {
  unsigned idx = blockIdx.x * 256u + threadIdx.x;
  if (idx >= (unsigned)BB * LQ * (FD / 2)) return;
  unsigned p = idx & (FD / 2 - 1);
  unsigned bt = idx >> (LG2FD - 1);
  unsigned t = bt & (LQ - 1);
  unsigned h = p >> 6, j = p & 63;
  unsigned base = bt << LG2FD;
  float x0 = Qraw[base + h * HD + j];
  float x1 = Qraw[base + h * HD + j + HHALF];
  float theta = __powf(1000.0f, -((float)(2 * j) / (float)HD));
  float s, c;
  __sincosf((float)t * theta, &s, &c);
  Qrope[base + h * HD + j]         = (__bf16)(x0 * c - x1 * s);
  Qrope[base + h * HD + j + HHALF] = (__bf16)(x1 * c + x0 * s);
}

// ---------------------------------------------------------------------------
// 4) C[M,FD] = A[M,FD] * B[FD,FD]  (f32 in/out, bf16 WMMA, f32 accumulate)
//    N = K = FD are compile-time -> all index math is 32-bit shift/add.
//    Block: 128 threads = 4 waves; block tile 64x64; wave tile 32x32; Ktile 32
// ---------------------------------------------------------------------------
__global__ __launch_bounds__(128) void gemm_bf16_kernel(
    const float* __restrict__ Aa, const float* __restrict__ Bb,
    float* __restrict__ Cc) {
  __shared__ __align__(16) __bf16 As[64][40];  // [m][k], 80B row (conflict-free)
  __shared__ __align__(16) __bf16 Bs[64][40];  // [n][k] (B staged transposed)

  unsigned tid = threadIdx.x;
  unsigned wave = tid >> 5, lane = tid & 31;
  unsigned hi = lane >> 4, l15 = lane & 15;
  unsigned wm = (wave & 1) * 32;
  unsigned wn = (wave >> 1) * 32;
  unsigned bm = blockIdx.y * 64;
  unsigned bn = blockIdx.x * 64;

  f32x8 acc[2][2] = {};

  for (unsigned k0 = 0; k0 < FD; k0 += 32) {
    // stage A (64x32): 4 x float4 per thread (global_load_b128, ds_store_b64)
#pragma unroll
    for (int it = 0; it < 4; ++it) {
      unsigned g = it * 512 + tid * 4;
      unsigned m = g >> 5, k = g & 31;
      const float4 v = *(const float4*)(Aa + ((bm + m) << LG2FD) + k0 + k);
      bf16x4 w = {(__bf16)v.x, (__bf16)v.y, (__bf16)v.z, (__bf16)v.w};
      *(bf16x4*)(&As[m][k]) = w;
    }
    // stage B transposed (n-major): float4 on n, scalar b16 stores
#pragma unroll
    for (int it = 0; it < 4; ++it) {
      unsigned g = it * 512 + tid * 4;
      unsigned n = g & 63, kk = g >> 6;
      const float4 v = *(const float4*)(Bb + ((k0 + kk) << LG2FD) + bn + n);
      Bs[n + 0][kk] = (__bf16)v.x;
      Bs[n + 1][kk] = (__bf16)v.y;
      Bs[n + 2][kk] = (__bf16)v.z;
      Bs[n + 3][kk] = (__bf16)v.w;
    }
    __syncthreads();

    bf16x16 af[2], bf[2];
#pragma unroll
    for (int s = 0; s < 2; ++s) {
      unsigned m = wm + s * 16 + l15;
      unsigned n = wn + s * 16 + l15;
#pragma unroll
      for (int i = 0; i < 16; ++i) {
        af[s][i] = As[m][kmap(i, hi)];
        bf[s][i] = Bs[n][kmap(i, hi)];
      }
    }
#pragma unroll
    for (int s = 0; s < 2; ++s)
#pragma unroll
      for (int t = 0; t < 2; ++t)
        acc[s][t] = wmma_bf16(af[s], bf[t], acc[s][t]);
    __syncthreads();
  }

  // C/D layout: lanes 0-15 -> rows r, lanes 16-31 -> rows r+8; col = lane&15
#pragma unroll
  for (int s = 0; s < 2; ++s)
#pragma unroll
    for (int t = 0; t < 2; ++t)
#pragma unroll
      for (int r = 0; r < 8; ++r) {
        unsigned row = bm + wm + s * 16 + r + hi * 8;
        unsigned col = bn + wn + t * 16 + l15;
        Cc[(row << LG2FD) + col] = acc[s][t][r];
      }
}

// ---------------------------------------------------------------------------
// 5) Flash attention: one wave owns a 16-row Q tile of one (b, h).
//    All operands pre-converted bf16; 32-bit offsets fold into VMEM offsets.
//    Per 32-key chunk: 8 score WMMAs (hd=128) + online softmax + 8 P*V WMMAs.
// ---------------------------------------------------------------------------
__global__ __launch_bounds__(128) void attention_kernel(
    const __bf16* __restrict__ Q, const __bf16* __restrict__ Krope,
    const __bf16* __restrict__ Vsep, const int* __restrict__ mask,
    float* __restrict__ attn) {
  __shared__ __align__(16) __bf16 Pl[4][16][40];  // per-wave P staging (16x32)

  unsigned lane = threadIdx.x & 31;
  unsigned wave = threadIdx.x >> 5;
  unsigned w = blockIdx.x * 4 + wave;
  unsigned qt = w & 63;            // LQ/16 = 64 tiles
  unsigned h = (w >> 6) & 15;
  unsigned b = w >> 10;
  unsigned hi = lane >> 4, l15 = lane & 15;
  unsigned qbase = qt * 16;

  // 64-bit base pointers hoisted once; everything below is 32-bit offsets.
  const __bf16* Kb = Krope + ((b * LKK) << LG2FD) + h * HD;   // [kpos<<11 + d]
  const __bf16* Vb = Vsep + ((b * LKK) << LG2FD) + h * HD;    // [kpos<<11 + d]
  const int* Mb = mask + ((b * LQ + qbase) << 11);            // [row<<11 + kpos]
  float* Ob = attn + ((b * LQ + qbase) << LG2FD) + h * HD;    // [row<<11 + d]

  // Q A-fragments: 4 chunks of K=32 covering hd=128; lane row = qbase + l15
  bf16x16 qf[4];
  {
    const __bf16* qrow = Q + ((b * LQ + qbase + l15) << LG2FD) + h * HD;
#pragma unroll
    for (int c = 0; c < 4; ++c)
#pragma unroll
      for (int i = 0; i < 16; ++i)
        qf[c][i] = qrow[c * 32 + kmap(i, hi)];
  }

  f32x8 oacc[8] = {};
  float mrow[8], lrow[8];
#pragma unroll
  for (int r = 0; r < 8; ++r) { mrow[r] = -3.0e38f; lrow[r] = 0.f; }
  const float sc = 0.08838834764831845f;  // 1/sqrt(128)

  for (unsigned kc = 0; kc < LKK; kc += 32) {
    // prefetch next chunk's K rows (gfx1250 global_prefetch_b8)
    if (kc + 32 < LKK) {
      const __bf16* nk = Kb + ((kc + 32 + l15) << LG2FD);
      __builtin_prefetch(nk, 0, 1);
      __builtin_prefetch(nk + (16u << LG2FD), 0, 1);
    }
    // ---- scores: two 16x16 N-tiles over this 32-key chunk ----
    f32x8 s[2] = {};
#pragma unroll
    for (int t = 0; t < 2; ++t) {
      unsigned koff = (kc + t * 16 + l15) << LG2FD;
#pragma unroll
      for (int c = 0; c < 4; ++c) {
        bf16x16 kf;
#pragma unroll
        for (int i = 0; i < 16; ++i)
          kf[i] = Kb[koff + c * 32 + kmap(i, hi)];
        s[t] = wmma_bf16(qf[c], kf, s[t]);
      }
    }
    // ---- scale + mask + online softmax (row reductions over 16-lane half) ----
    float p[2][8];
#pragma unroll
    for (int r = 0; r < 8; ++r) {
      unsigned roff = (r + hi * 8) << 11;
      float sv[2], chmax = -3.0e38f;
#pragma unroll
      for (int t = 0; t < 2; ++t) {
        float v = s[t][r] * sc;
        v = (Mb[roff + kc + t * 16 + l15] > 0) ? v : -3.0e38f;
        sv[t] = v;
        chmax = fmaxf(chmax, v);
      }
#pragma unroll
      for (int d = 1; d < 16; d <<= 1)
        chmax = fmaxf(chmax, __shfl_xor(chmax, d, 32));
      float mnew = fmaxf(mrow[r], chmax);
      float resc = __expf(mrow[r] - mnew);
      float chsum = 0.f;
#pragma unroll
      for (int t = 0; t < 2; ++t) {
        float pv = __expf(sv[t] - mnew);
        p[t][r] = pv;
        chsum += pv;
      }
#pragma unroll
      for (int d = 1; d < 16; d <<= 1) chsum += __shfl_xor(chsum, d, 32);
      lrow[r] = lrow[r] * resc + chsum;
      mrow[r] = mnew;
#pragma unroll
      for (int nt = 0; nt < 8; ++nt) oacc[nt][r] *= resc;
    }
    // ---- re-layout P via wave-private LDS into an A-fragment (16x32) ----
#pragma unroll
    for (int r = 0; r < 8; ++r)
#pragma unroll
      for (int t = 0; t < 2; ++t)
        Pl[wave][r + hi * 8][t * 16 + l15] = (__bf16)p[t][r];
    bf16x16 pf;
#pragma unroll
    for (int i = 0; i < 16; ++i)
      pf[i] = Pl[wave][l15][kmap(i, hi)];
    // ---- P * V: 8 output N-tiles over hd ----
#pragma unroll
    for (int nt = 0; nt < 8; ++nt) {
      bf16x16 vf;
      unsigned col = nt * 16 + l15;
#pragma unroll
      for (int i = 0; i < 16; ++i)
        vf[i] = Vb[((kc + kmap(i, hi)) << LG2FD) + col];
      oacc[nt] = wmma_bf16(pf, vf, oacc[nt]);
    }
  }

  // ---- normalize + store ----
#pragma unroll
  for (int nt = 0; nt < 8; ++nt)
#pragma unroll
    for (int r = 0; r < 8; ++r)
      Ob[((r + hi * 8) << LG2FD) + nt * 16 + l15] = oacc[nt][r] / lrow[r];
}

// ---------------------------------------------------------------------------
extern "C" void kernel_launch(void* const* d_in, const int* in_sizes, int n_in,
                              void* d_out, int out_size, void* d_ws, size_t ws_size,
                              hipStream_t stream) {
  (void)in_sizes; (void)n_in; (void)out_size; (void)ws_size;
  const float* kv       = (const float*)d_in[0];
  const float* q        = (const float*)d_in[1];
  const float* kv_cache = (const float*)d_in[2];
  const float* Wq       = (const float*)d_in[3];
  const float* Wk       = (const float*)d_in[4];
  const float* Wv       = (const float*)d_in[5];
  const float* Wo       = (const float*)d_in[6];
  const int*   mask     = (const int*)d_in[7];

  float* logits = (float*)d_out;                       // [B, Lq, F]
  float* kv_out = logits + (long)BB * LQ * FD;         // [B, Lk, F, 2]

  // Workspace layout (f32 region first, then bf16 region, then f32 attn)
  float*  ws    = (float*)d_ws;
  float*  kn    = ws;                                  // [B, F]        f32
  float*  vn    = kn + (long)BB * FD;                  // [B, F]        f32
  float*  Qraw  = vn + (long)BB * FD;                  // [B*Lq, F]     f32
  __bf16* Qrope = (__bf16*)(Qraw + (long)BB * LQ * FD);       // [B*Lq, F]
  __bf16* Krope = Qrope + (long)BB * LQ * FD;                 // [B*Lk, F]
  __bf16* Vsep  = Krope + (long)BB * LKK * FD;                // [B*Lk, F]
  float*  attn  = (float*)(Vsep + (long)BB * LKK * FD);       // [B*Lq, F] f32

  // 1) new-token K/V projection
  kv_proj_kernel<<<(BB * FD + 255) / 256, 256, 0, stream>>>(kv, Wk, Wv, kn, vn);

  // 2) cache concat (d_out) + bf16 RoPE'd K + bf16 V plane (ws)
  unsigned nck = (unsigned)BB * LKK * (FD / 2);
  concat_rope_k_kernel<<<(nck + 255) / 256, 256, 0, stream>>>(
      kv_cache, kn, vn, kv_out, Krope, Vsep);

  // 3) Q projection (WMMA GEMM), then RoPE -> bf16
  dim3 g1(FD / 64, (BB * LQ) / 64);
  gemm_bf16_kernel<<<g1, 128, 0, stream>>>(q, Wq, Qraw);
  unsigned nrq = (unsigned)BB * LQ * (FD / 2);
  rope_q_kernel<<<(nrq + 255) / 256, 256, 0, stream>>>(Qraw, Qrope);

  // 4) flash attention (WMMA)
  attention_kernel<<<BB * NH * (LQ / 16) / 4, 128, 0, stream>>>(
      Qrope, Krope, Vsep, mask, attn);

  // 5) output projection (WMMA GEMM) -> logits
  gemm_bf16_kernel<<<g1, 128, 0, stream>>>(attn, Wo, logits);
}
